// LSTM_35107062677782
// MI455X (gfx1250) — matrix operand order repaired and checked by reading, not compiled, not run on hardware
//
#include <hip/hip_runtime.h>
#include <hip/hip_bf16.h>

// Problem constants (match reference)
constexpr int Tn = 512;
constexpr int Bn = 64;
constexpr int Dn = 1024;   // == Hn, so both layers share GEMM shapes
constexpr int Hn = 1024;
constexpr int G4 = 4096;   // 4*H
constexpr int Ln = 2;

typedef __attribute__((ext_vector_type(16))) __bf16 v16bf;
typedef __attribute__((ext_vector_type(8)))  __bf16 v8bf;
typedef __attribute__((ext_vector_type(8)))  float  v8f;

__device__ __forceinline__ v8f wmma_bf16(v16bf a, v16bf b, v8f c) {
  // D = A(16x32 bf16) * B(32x16 bf16) + C(16x16 f32)
  return __builtin_amdgcn_wmma_f32_16x16x32_bf16(
      /*neg_a=*/false, a, /*neg_b=*/false, b,
      /*c_mod=*/(short)0, c, /*reuse_a=*/false, /*reuse_b=*/false);
}

__device__ __forceinline__ float sigmoidf_fast(float x) {
  return 1.0f / (1.0f + __expf(-x));
}

// ---------------------------------------------------------------------------
// fp32 -> bf16 conversion / fills
// ---------------------------------------------------------------------------
__global__ void cvt_f32_to_bf16(const float* __restrict__ in,
                                __bf16* __restrict__ out, size_t n) {
  size_t i = (size_t)blockIdx.x * blockDim.x + threadIdx.x;
  size_t stride = (size_t)gridDim.x * blockDim.x;
  for (; i < n; i += stride) out[i] = (__bf16)in[i];
}

__global__ void fill_zero_bf16(__bf16* __restrict__ p, size_t n) {
  size_t i = (size_t)blockIdx.x * blockDim.x + threadIdx.x;
  size_t stride = (size_t)gridDim.x * blockDim.x;
  for (; i < n; i += stride) p[i] = (__bf16)0.0f;
}

__global__ void fill_zero_u32(unsigned* __restrict__ p, size_t n) {
  size_t i = (size_t)blockIdx.x * blockDim.x + threadIdx.x;
  size_t stride = (size_t)gridDim.x * blockDim.x;
  for (; i < n; i += stride) p[i] = 0u;
}

// ---------------------------------------------------------------------------
// Big input-projection GEMM: C[M=T*B, 4096] = A[M, 1024] * Wih^T
// Wih is row-major [4096, 1024] so B-fragment columns are contiguous W rows.
// Each wave: 32 rows x 64 cols = 2 A-fragments x 4 B-fragments -> 8 WMMAs
// per K-step (8 WMMA : 12 b128 loads).
// Grid: (Mtiles=1024, 8), block = 256 (8 waves); wave -> one 32x64 tile.
// ---------------------------------------------------------------------------
__global__ __launch_bounds__(256) void gemm_xproj(
    const __bf16* __restrict__ A,   // [M, 1024] bf16 row-major
    const __bf16* __restrict__ W,   // [4096, 1024] bf16 row-major
    float* __restrict__ C)          // [M, 4096] f32
{
  const int lane   = threadIdx.x & 31;
  const int wave   = threadIdx.x >> 5;
  const int mtile  = blockIdx.x;                 // 0..1023  (32-row tiles)
  const int ntile  = blockIdx.y * 8 + wave;      // 0..63    (64-col tiles)
  const int rowBase = mtile * 32;
  const int colBase = ntile * 64;

  const int lrow = lane & 15;
  const int khA  = (lane >> 4) * 16;   // A: K-half select (elements)
  const int khB  = (lane >> 4) * 16;   // B: K-half select (elements)

  const __bf16* arow0 = A + (size_t)(rowBase + lrow) * Dn;
  const __bf16* arow1 = A + (size_t)(rowBase + 16 + lrow) * Dn;
  const __bf16* wrow[4];
#pragma unroll
  for (int s = 0; s < 4; ++s)
    wrow[s] = W + (size_t)(colBase + s * 16 + lrow) * Dn;

  v8f acc[2][4] = {};
  for (int k0 = 0; k0 < Dn; k0 += 32) {
    union { v16bf v; v8bf h[2]; } a0, a1;
    a0.h[0] = *(const v8bf*)(arow0 + k0 + khA);
    a0.h[1] = *(const v8bf*)(arow0 + k0 + 8 + khA);
    a1.h[0] = *(const v8bf*)(arow1 + k0 + khA);
    a1.h[1] = *(const v8bf*)(arow1 + k0 + 8 + khA);
#pragma unroll
    for (int s = 0; s < 4; ++s) {
      v16bf b = *(const v16bf*)(wrow[s] + k0 + khB);
      acc[0][s] = wmma_bf16(a0.v, b, acc[0][s]);
      acc[1][s] = wmma_bf16(a1.v, b, acc[1][s]);
    }
  }

#pragma unroll
  for (int m = 0; m < 2; ++m) {
#pragma unroll
    for (int s = 0; s < 4; ++s) {
      const int col = colBase + s * 16 + lrow;
#pragma unroll
      for (int r = 0; r < 8; ++r) {
        const int row = rowBase + m * 16 + r + (lane >> 4) * 8;
        C[(size_t)row * G4 + col] = acc[m][s][r];
      }
    }
  }
}

// ---------------------------------------------------------------------------
// Device-scope grid barrier (32 co-resident blocks)
// ---------------------------------------------------------------------------
__device__ __forceinline__ void grid_sync(unsigned* cnt, unsigned* gen,
                                          unsigned nblocks) {
  __syncthreads();
  if (threadIdx.x == 0) {
    __threadfence();  // release prior h stores to device scope
    unsigned g = __hip_atomic_load(gen, __ATOMIC_ACQUIRE,
                                   __HIP_MEMORY_SCOPE_AGENT);
    unsigned arrived = __hip_atomic_fetch_add(cnt, 1u, __ATOMIC_ACQ_REL,
                                              __HIP_MEMORY_SCOPE_AGENT);
    if (arrived == nblocks - 1) {
      __hip_atomic_store(cnt, 0u, __ATOMIC_RELAXED, __HIP_MEMORY_SCOPE_AGENT);
      __hip_atomic_fetch_add(gen, 1u, __ATOMIC_ACQ_REL,
                             __HIP_MEMORY_SCOPE_AGENT);
    } else {
      while (__hip_atomic_load(gen, __ATOMIC_ACQUIRE,
                               __HIP_MEMORY_SCOPE_AGENT) == g) {
        __builtin_amdgcn_s_sleep(1);
      }
    }
  }
  __syncthreads();
}

// ---------------------------------------------------------------------------
// Persistent recurrent kernel: runs all 512 timesteps of one layer.
// 256 waves total (32 blocks x 8 waves); wave owns a 16(batch) x 16(hidden)
// tile and accumulates all 4 gate tiles (i,f,g,o). Cell state c lives in
// registers for the whole sequence. The per-step xproj bias tile is loaded
// BEFORE the K-loop so its L2 latency hides behind the GEMM. h ping-pongs
// through device memory with an atomic grid barrier per step.
// ---------------------------------------------------------------------------
__global__ __launch_bounds__(256) void lstm_seq(
    __bf16* __restrict__ h0,            // [64*1024] bf16, zeroed (t even: prev)
    __bf16* __restrict__ h1,            // [64*1024] bf16 (t odd: prev)
    const __bf16* __restrict__ Whh,     // [4096, 1024] bf16 row-major
    const float* __restrict__ xproj,    // [T, 64, 4096] f32
    __bf16* __restrict__ hseq_out,      // [T, 64, 1024] bf16 (layer0) or null
    float* __restrict__ f32_out,        // [T, 64, 1024] f32 (layer1) or null
    unsigned* __restrict__ cnt, unsigned* __restrict__ gen)
{
  const int lane  = threadIdx.x & 31;
  const int wave  = threadIdx.x >> 5;
  const int gwave = blockIdx.x * 8 + wave;    // 0..255
  const int rowBase = (gwave >> 6) * 16;      // 4 batch-row tiles
  const int colBase = (gwave & 63) * 16;      // 64 hidden-col tiles

  const int lrow = lane & 15;
  const int khA  = (lane >> 4) * 16;
  const int khB  = (lane >> 4) * 16;
  const int col  = colBase + lrow;

  const __bf16* wrow[4];
#pragma unroll
  for (int g = 0; g < 4; ++g)
    wrow[g] = Whh + (size_t)(g * Hn + colBase + lrow) * Hn;

  float creg[8] = {};  // cell state, register-resident across all timesteps

  for (int t = 0; t < Tn; ++t) {
    const __bf16* hprev = (t & 1) ? h1 : h0;
    __bf16*       hnext = (t & 1) ? h0 : h1;
    const float*  xp    = xproj + (size_t)t * (Bn * G4);

    // Hoist the xproj bias tile loads: independent of the GEMM, so issue
    // them first and let the K-loop cover their latency.
    float xpv[4][8];
#pragma unroll
    for (int g = 0; g < 4; ++g) {
#pragma unroll
      for (int r = 0; r < 8; ++r) {
        const int row = rowBase + r + (lane >> 4) * 8;
        xpv[g][r] = xp[(size_t)row * G4 + g * Hn + col];
      }
    }

    // gates[16x16 tile, 4 gate blocks] = h_prev @ Whh^T
    v8f acc[4] = {};
    const __bf16* arow = hprev + (size_t)(rowBase + lrow) * Hn;
    for (int k0 = 0; k0 < Hn; k0 += 32) {
      union { v16bf v; v8bf h[2]; } a;
      a.h[0] = *(const v8bf*)(arow + k0 + khA);
      a.h[1] = *(const v8bf*)(arow + k0 + 8 + khA);
#pragma unroll
      for (int g = 0; g < 4; ++g) {
        v16bf b = *(const v16bf*)(wrow[g] + k0 + khB);
        acc[g] = wmma_bf16(a.v, b, acc[g]);
      }
    }

    // gate nonlinearities + cell/hidden update, all in registers
#pragma unroll
    for (int r = 0; r < 8; ++r) {
      const int row = rowBase + r + (lane >> 4) * 8;
      float gi = acc[0][r] + xpv[0][r];
      float gf = acc[1][r] + xpv[1][r];
      float gg = acc[2][r] + xpv[2][r];
      float go = acc[3][r] + xpv[3][r];
      gi = sigmoidf_fast(gi);
      gf = sigmoidf_fast(gf);
      go = sigmoidf_fast(go);
      gg = tanhf(gg);
      const float cn = gf * creg[r] + gi * gg;
      creg[r] = cn;
      const float hv = go * tanhf(cn);
      const size_t hi = (size_t)row * Hn + col;
      hnext[hi] = (__bf16)hv;
      const size_t oi = (size_t)t * (Bn * Hn) + hi;
      if (hseq_out) hseq_out[oi] = (__bf16)hv;
      else          f32_out[oi]  = hv;
    }

    grid_sync(cnt, gen, gridDim.x);
  }
}

// ---------------------------------------------------------------------------
// Host launcher
// ---------------------------------------------------------------------------
extern "C" void kernel_launch(void* const* d_in, const int* in_sizes, int n_in,
                              void* d_out, int out_size, void* d_ws, size_t ws_size,
                              hipStream_t stream) {
  const float* x   = (const float*)d_in[0];  // [T, B, D]
  const float* Wih = (const float*)d_in[1];  // [L, 4H, H]
  const float* Whh = (const float*)d_in[2];  // [L, 4H, H]
  float* out = (float*)d_out;                // [T, B, H]

  // Workspace carve-up (all offsets 256B aligned)
  char* ws = (char*)d_ws;
  size_t off = 0;
  auto carve = [&](size_t bytes) -> void* {
    void* p = ws + off;
    off = (off + bytes + 255) & ~(size_t)255;
    return p;
  };
  __bf16* xbf   = (__bf16*)carve((size_t)Tn * Bn * Dn * sizeof(__bf16)); // 64MB
  __bf16* hseq  = (__bf16*)carve((size_t)Tn * Bn * Hn * sizeof(__bf16)); // 64MB
  __bf16* wihb  = (__bf16*)carve((size_t)Ln * G4 * Hn * sizeof(__bf16)); // 16MB
  __bf16* whhb  = (__bf16*)carve((size_t)Ln * G4 * Hn * sizeof(__bf16)); // 16MB
  float*  xproj = (float*)carve((size_t)Tn * Bn * G4 * sizeof(float));   // 512MB
  __bf16* hbuf0 = (__bf16*)carve((size_t)Bn * Hn * sizeof(__bf16));
  __bf16* hbuf1 = (__bf16*)carve((size_t)Bn * Hn * sizeof(__bf16));
  unsigned* bar = (unsigned*)carve(2 * sizeof(unsigned)); // [cnt, gen]

  // Convert activations and weights to bf16 once
  const size_t nx = (size_t)Tn * Bn * Dn;
  const size_t nw = (size_t)Ln * G4 * Hn;
  cvt_f32_to_bf16<<<2048, 256, 0, stream>>>(x,   xbf,  nx);
  cvt_f32_to_bf16<<<2048, 256, 0, stream>>>(Wih, wihb, nw);
  cvt_f32_to_bf16<<<2048, 256, 0, stream>>>(Whh, whhb, nw);

  for (int l = 0; l < Ln; ++l) {
    const __bf16* in_bf = (l == 0) ? xbf : hseq;
    const __bf16* wih_l = wihb + (size_t)l * G4 * Hn;
    const __bf16* whh_l = whhb + (size_t)l * G4 * Hn;

    // Phase 1: x_proj = in @ Wih^T  (big parallel GEMM)
    gemm_xproj<<<dim3(1024, 8), 256, 0, stream>>>(in_bf, wih_l, xproj);

    // Reset h0 and barrier state
    fill_zero_bf16<<<64, 256, 0, stream>>>(hbuf0, (size_t)Bn * Hn);
    fill_zero_u32<<<1, 32, 0, stream>>>(bar, 2);

    // Phase 2: persistent recurrent kernel over all 512 timesteps
    lstm_seq<<<32, 256, 0, stream>>>(
        hbuf0, hbuf1, whh_l, xproj,
        (l == 0) ? hseq : nullptr,
        (l == Ln - 1) ? out : nullptr,
        bar + 0, bar + 1);
  }
}
